// EncoderLayer_82566451299290
// MI455X (gfx1250) — matrix operand order repaired and checked by reading, not compiled, not run on hardware
//
#include <hip/hip_runtime.h>

#define DEVINL __device__ __forceinline__

typedef __attribute__((ext_vector_type(16))) __bf16 v16bf;
typedef __attribute__((ext_vector_type(8)))  float  v8f;
typedef __attribute__((ext_vector_type(8)))  unsigned short v8us;
typedef __attribute__((ext_vector_type(16))) unsigned short v16us;
typedef __attribute__((ext_vector_type(4)))  int v4i;

constexpr int B = 8, N = 358, T = 36, D = 128, SEG = 12;
constexpr int LDY = 144;   // LDS row stride (ushorts) for 48x128 row-major tiles
constexpr int VST = 56;    // LDS stride for transposed V (128 ch x 48 keys pad)
constexpr int NP  = 384;   // padded node count
constexpr int MS  = 392;   // LDS stride for channel-major node slab (384+8)
constexpr int GS  = 136;   // LDS stride for staged G slab (128+8)

// ---- workspace layout (bytes) ----
constexpr size_t OFF_WQP = 0;                       // 3*128*128*2 = 98304
constexpr size_t OFF_WKP = 98304;
constexpr size_t OFF_FCP = 196608;
constexpr size_t OFF_PJP = 229376;
constexpr size_t OFF_THP = 262144;
constexpr size_t OFF_ADJ = 294912;                  // 384*384*2 = 294912
constexpr size_t OFF_Y2  = 589824;                  // B*N*T*D*2 = 26394624
constexpr size_t OFF_GB  = 589824 + 26394624;       // B*T*384*128*2 = 28311552

#if defined(__has_builtin)
#if __has_builtin(__builtin_amdgcn_global_load_async_to_lds_b128)
#define HAVE_ASYNC_LDS 1
#endif
#endif

#ifdef HAVE_ASYNC_LDS
typedef __attribute__((address_space(1))) v4i* gv4i_ptr;
typedef __attribute__((address_space(3))) v4i* lv4i_ptr;
DEVINL void async_wait0() {
#if __has_builtin(__builtin_amdgcn_s_wait_asynccnt)
  __builtin_amdgcn_s_wait_asynccnt(0);
#else
  asm volatile("s_wait_asynccnt 0" ::: "memory");
#endif
}
#endif

DEVINL unsigned short f2bf(float f) {
  // native RNE f32->bf16 convert (lowers to v_cvt_pk_bf16_f32 on gfx1250)
  return __builtin_bit_cast(unsigned short, (__bf16)f);
}
DEVINL v16bf mkfrag(v8us lo, v8us hi) {
  v16us t = __builtin_shufflevector(lo, hi, 0,1,2,3,4,5,6,7,8,9,10,11,12,13,14,15);
  return __builtin_bit_cast(v16bf, t);
}
DEVINL v8f wmma_bf(v16bf a, v16bf b, v8f c) {
  return __builtin_amdgcn_wmma_f32_16x16x32_bf16(false, a, false, b, (short)0, c, false, false);
}
DEVINL unsigned long long pack4(unsigned short a, unsigned short b,
                                unsigned short c, unsigned short d) {
  return (unsigned long long)a | ((unsigned long long)b << 16) |
         ((unsigned long long)c << 32) | ((unsigned long long)d << 48);
}

// ---------------------------------------------------------------------------
// Kernel 0: pack weights to bf16 WMMA-friendly layouts
// conv packs: [k][o][i]; fc/proj/theta: transposed (o,i); adj: 384x384 padded
// ---------------------------------------------------------------------------
__global__ void __launch_bounds__(256)
pack_kernel(const float* __restrict__ cqw, const float* __restrict__ ckw,
            const float* __restrict__ fcw, const float* __restrict__ pjw,
            const float* __restrict__ thw, const float* __restrict__ adj,
            unsigned short* __restrict__ wqp, unsigned short* __restrict__ wkp,
            unsigned short* __restrict__ fcp, unsigned short* __restrict__ pjp,
            unsigned short* __restrict__ thp, unsigned short* __restrict__ adjp) {
  int i = blockIdx.x * 256 + threadIdx.x;
  if (i < 3 * 128 * 128) {
    int k = i / (128 * 128), r = i % (128 * 128);
    int o = r / 128, ii = r % 128;
    wqp[i] = f2bf(cqw[(o * 128 + ii) * 3 + k]);
    wkp[i] = f2bf(ckw[(o * 128 + ii) * 3 + k]);
  }
  if (i < 128 * 128) {
    int o = i / 128, ii = i % 128;
    fcp[i] = f2bf(fcw[ii * 128 + o]);
    pjp[i] = f2bf(pjw[ii * 128 + o]);
    thp[i] = f2bf(thw[ii * 128 + o]);
  }
  if (i < NP * NP) {
    int n = i / NP, m = i % NP;
    adjp[i] = (n < N && m < N) ? f2bf(adj[n * N + m]) : (unsigned short)0;
  }
}

// ---------------------------------------------------------------------------
// Kernel 1: LN1 + conv-QK + V + attention + proj + residual  (block per (b,n))
// ---------------------------------------------------------------------------
__global__ void __launch_bounds__(256)
sublayer1_kernel(const float* __restrict__ x,
                 const float* __restrict__ cqb, const float* __restrict__ ckb,
                 const float* __restrict__ fcb, const float* __restrict__ pjb,
                 const float* __restrict__ ln1g, const float* __restrict__ ln1b,
                 const unsigned short* __restrict__ wqp,
                 const unsigned short* __restrict__ wkp,
                 const unsigned short* __restrict__ fcp,
                 const unsigned short* __restrict__ pjp,
                 float* __restrict__ x1) {
  __shared__ unsigned short ylds[48 * LDY];
  __shared__ unsigned short qlds[48 * LDY];
  __shared__ unsigned short klds[48 * LDY];
  __shared__ unsigned short vldt[128 * VST];   // V transposed: [channel][key]
  __shared__ unsigned short clds[48 * LDY];

  const int tid  = threadIdx.x;
  const int lane = tid & 31;
  const int wave = tid >> 5;
  const int lh   = lane >> 4;      // lane half (0/1)
  const int l15  = lane & 15;
  const int kb8  = 8 * lh;         // A-frag K base
  const int bn   = blockIdx.x;
  const float* xrow0 = x + (size_t)bn * T * D;

  // ---- Phase A: LN1 -> ylds bf16; zero rows 36..47 ----
  for (int row = wave; row < T; row += 8) {
    float4 v = *(const float4*)(xrow0 + row * D + lane * 4);
    float s = v.x + v.y + v.z + v.w;
#pragma unroll
    for (int m = 16; m >= 1; m >>= 1) s += __shfl_xor(s, m, 32);
    float mean = s * (1.f / 128.f);
    float d0 = v.x - mean, d1 = v.y - mean, d2 = v.z - mean, d3 = v.w - mean;
    float q = d0 * d0 + d1 * d1 + d2 * d2 + d3 * d3;
#pragma unroll
    for (int m = 16; m >= 1; m >>= 1) q += __shfl_xor(q, m, 32);
    float rstd = rsqrtf(q * (1.f / 128.f) + 1e-5f);
    int c = lane * 4;
    unsigned long long pk =
        pack4(f2bf(d0 * rstd * ln1g[c + 0] + ln1b[c + 0]),
              f2bf(d1 * rstd * ln1g[c + 1] + ln1b[c + 1]),
              f2bf(d2 * rstd * ln1g[c + 2] + ln1b[c + 2]),
              f2bf(d3 * rstd * ln1g[c + 3] + ln1b[c + 3]));
    *(unsigned long long*)(ylds + row * LDY + c) = pk;
  }
  for (int i = tid; i < 12 * LDY; i += 256)
    ylds[(36 + i / LDY) * LDY + (i % LDY)] = 0;
  __syncthreads();

  // A-frag from an LDS bf16 row-major buffer
  auto loadA = [&](const unsigned short* buf, int row, int c32) -> v16bf {
    const unsigned short* p = buf + row * LDY + c32 + kb8;
    return mkfrag(*(const v8us*)p, *(const v8us*)(p + 16));
  };
  // B-frag from packed (o,i) bf16 weight
  auto loadB = [&](const unsigned short* w, int ocol, int c32) -> v16bf {
    const unsigned short* p = w + ocol * 128 + c32 + 8 * lh;
    return mkfrag(*(const v8us*)p, *(const v8us*)(p + 16));
  };

  // ---- Phase B: Q/K (3-tap conv GEMM) and V (fc GEMM); 72 tiles / 8 waves
  for (int tI = wave; tI < 72; tI += 8) {
    int mat = tI / 24, rem = tI % 24, mt = rem >> 3, nt = rem & 7;
    int ocol = nt * 16 + l15;
    int trow = mt * 16 + l15;
    v8f acc = {};
    if (mat < 2) {
      const unsigned short* wp = (mat == 0) ? wqp : wkp;
#pragma unroll
      for (int ks = 0; ks < 3; ++ks) {
        int sr = trow + ks - 1;
        bool valid = (sr >= 0) && (sr < 48) && (sr / SEG == trow / SEG);
        int arow = valid ? sr : 44;   // row 44 guaranteed zero
#pragma unroll
        for (int c = 0; c < 4; ++c) {
          v16bf a = loadA(ylds, arow, c * 32);
          v16bf b = loadB(wp + ks * 128 * 128, ocol, c * 32);
          acc = wmma_bf(a, b, acc);
        }
      }
      float bias = (mat == 0) ? cqb[ocol] : ckb[ocol];
      unsigned short* dst = (mat == 0) ? qlds : klds;
#pragma unroll
      for (int r = 0; r < 8; ++r)
        dst[(mt * 16 + r + 8 * lh) * LDY + ocol] = f2bf(acc[r] + bias);
    } else {
#pragma unroll
      for (int c = 0; c < 4; ++c) {
        v16bf a = loadA(ylds, trow, c * 32);
        v16bf b = loadB(fcp, ocol, c * 32);
        acc = wmma_bf(a, b, acc);
      }
      float bias = fcb[ocol];
      v8us vv;
#pragma unroll
      for (int r = 0; r < 8; ++r) vv[r] = f2bf(acc[r] + bias);
      // transposed store: contiguous keys per channel, single 16B ds store
      *(v8us*)(vldt + ocol * VST + mt * 16 + 8 * lh) = vv;
    }
  }
  __syncthreads();

  // ---- Phase C: attention, one head per wave. S^T = K @ Q^T. ----
  {
    const int h = wave, ch0 = h * 16;
    v8us z8 = {0, 0, 0, 0, 0, 0, 0, 0};
    v16bf Ak[3], Bq[3];
#pragma unroll
    for (int mt = 0; mt < 3; ++mt) {
      const unsigned short* p = klds + (mt * 16 + l15) * LDY + ch0 + kb8;
      Ak[mt] = mkfrag(*(const v8us*)p, z8);     // K pad: channels 16..31 = 0
    }
#pragma unroll
    for (int nt = 0; nt < 3; ++nt) {
      const unsigned short* p = qlds + (nt * 16 + l15) * LDY + ch0 + kb8;
      Bq[nt] = mkfrag(*(const v8us*)p, z8);      // Q pad: channels 16..31 = 0
    }
    v8f S[3][3];
#pragma unroll
    for (int mt = 0; mt < 3; ++mt)
#pragma unroll
      for (int nt = 0; nt < 3; ++nt) {
        v8f z = {};
        S[mt][nt] = wmma_bf(Ak[mt], Bq[nt], z);
      }
    // softmax over keys (rows of S^T): per-lane reduce + shfl_xor(16)
#pragma unroll
    for (int nt = 0; nt < 3; ++nt) {
      float mx = -1e30f;
#pragma unroll
      for (int mt = 0; mt < 3; ++mt)
#pragma unroll
        for (int r = 0; r < 8; ++r) {
          int key = mt * 16 + r + 8 * lh;
          float s = S[mt][nt][r] * 0.25f;      // 1/sqrt(HD)
          if (key >= T) s = -1e30f;
          S[mt][nt][r] = s;
          mx = fmaxf(mx, s);
        }
      mx = fmaxf(mx, __shfl_xor(mx, 16, 32));
      float ss = 0.f;
#pragma unroll
      for (int mt = 0; mt < 3; ++mt)
#pragma unroll
        for (int r = 0; r < 8; ++r) {
          float p = __expf(S[mt][nt][r] - mx);
          S[mt][nt][r] = p;
          ss += p;
        }
      ss += __shfl_xor(ss, 16, 32);
      float inv = 1.f / ss;
#pragma unroll
      for (int mt = 0; mt < 3; ++mt)
#pragma unroll
        for (int r = 0; r < 8; ++r) S[mt][nt][r] *= inv;
    }
    // V B-frags: contiguous b128 loads from transposed V
    v16bf Bv[2];
    {
      const unsigned short* pv = vldt + (ch0 + l15) * VST;
      Bv[0] = mkfrag(*(const v8us*)(pv + 8 * lh), *(const v8us*)(pv + 8 * lh + 16));
      Bv[1] = mkfrag(*(const v8us*)(pv + 32 + 8 * lh), z8);  // keys 48..63 = 0
    }
    // context: P(C-layout) fragments map directly onto A-layout
#pragma unroll
    for (int qt = 0; qt < 3; ++qt) {
      v16us a0, a1;
#pragma unroll
      for (int r = 0; r < 8; ++r) {
        a0[r]     = f2bf(S[0][qt][r]);
        a0[8 + r] = f2bf(S[1][qt][r]);
        a1[r]     = f2bf(S[2][qt][r]);
        a1[8 + r] = 0;
      }
      v8f acc = {};
      acc = wmma_bf(__builtin_bit_cast(v16bf, a0), Bv[0], acc);
      acc = wmma_bf(__builtin_bit_cast(v16bf, a1), Bv[1], acc);
#pragma unroll
      for (int r = 0; r < 8; ++r)
        clds[(qt * 16 + r + 8 * lh) * LDY + ch0 + l15] = f2bf(acc[r]);
    }
  }
  __syncthreads();

  // ---- Phase D: proj + bias + residual -> x1 (d_out) ----
  for (int tI = wave; tI < 24; tI += 8) {
    int mt = tI >> 3, nt = tI & 7;
    int ocol = nt * 16 + l15;
    v8f acc = {};
#pragma unroll
    for (int c = 0; c < 4; ++c) {
      v16bf a = loadA(clds, mt * 16 + l15, c * 32);
      v16bf b = loadB(pjp, ocol, c * 32);
      acc = wmma_bf(a, b, acc);
    }
    float bias = pjb[ocol];
#pragma unroll
    for (int r = 0; r < 8; ++r) {
      int row = mt * 16 + r + 8 * lh;
      if (row < T) {
        size_t idx = (size_t)bn * T * D + (size_t)row * D + ocol;
        x1[idx] = x[idx] + acc[r] + bias;
      }
    }
  }
}

// ---------------------------------------------------------------------------
// Kernel 2: LN2 + transpose to (B,T,N,D) bf16
// ---------------------------------------------------------------------------
__global__ void __launch_bounds__(256)
ln2_kernel(const float* __restrict__ x1, const float* __restrict__ g,
           const float* __restrict__ bb, unsigned short* __restrict__ y2t) {
  int row = blockIdx.x * 8 + (threadIdx.x >> 5);
  int lane = threadIdx.x & 31;
  if (row >= B * N * T) return;
  int t = row % T, bn = row / T, n = bn % N, b = bn / N;
  const float* xr = x1 + (size_t)row * D;
  float4 v = *(const float4*)(xr + lane * 4);
  float s = v.x + v.y + v.z + v.w;
#pragma unroll
  for (int m = 16; m >= 1; m >>= 1) s += __shfl_xor(s, m, 32);
  float mean = s * (1.f / 128.f);
  float d0 = v.x - mean, d1 = v.y - mean, d2 = v.z - mean, d3 = v.w - mean;
  float q = d0 * d0 + d1 * d1 + d2 * d2 + d3 * d3;
#pragma unroll
  for (int m = 16; m >= 1; m >>= 1) q += __shfl_xor(q, m, 32);
  float rstd = rsqrtf(q * (1.f / 128.f) + 1e-5f);
  int c = lane * 4;
  unsigned long long pk =
      pack4(f2bf(d0 * rstd * g[c + 0] + bb[c + 0]),
            f2bf(d1 * rstd * g[c + 1] + bb[c + 1]),
            f2bf(d2 * rstd * g[c + 2] + bb[c + 2]),
            f2bf(d3 * rstd * g[c + 3] + bb[c + 3]));
  *(unsigned long long*)(y2t + (((size_t)(b * T + t) * N) + n) * D + c) = pk;
}

// ---------------------------------------------------------------------------
// Kernel 3: G = adj @ y2 per (b,t). Channel-major LDS staging so B-frags are
// two contiguous ds_load_b128. A-frags stream packed adjacency from L2.
// ---------------------------------------------------------------------------
__global__ void __launch_bounds__(256)
gcn1_kernel(const unsigned short* __restrict__ adjp,
            const unsigned short* __restrict__ y2t,
            unsigned short* __restrict__ gbuf) {
  __shared__ unsigned short yst[128 * MS];   // [channel][node]
  const int tid = threadIdx.x, lane = tid & 31, wave = tid >> 5;
  const int lh = lane >> 4, l15 = lane & 15;
  const int bt = blockIdx.x;
  const unsigned short* ysrc = y2t + (size_t)bt * N * D;

  // stage node slab with transpose (zeros beyond N)
  for (int i = tid; i < NP * 16; i += 256) {
    int m = i >> 4, c8 = (i & 15) * 8;
    v8us val = {0, 0, 0, 0, 0, 0, 0, 0};
    if (m < N) val = *(const v8us*)(ysrc + (size_t)m * D + c8);
#pragma unroll
    for (int j = 0; j < 8; ++j) yst[(c8 + j) * MS + m] = val[j];
  }
  __syncthreads();

  const int nt = wave, ch = nt * 16 + l15;
  unsigned short* gdst = gbuf + (size_t)bt * NP * D;
  for (int g0 = 0; g0 < 24; g0 += 4) {
    __builtin_prefetch(adjp + (size_t)(g0 * 16 + l15) * NP, 0, 1);
    v8f acc[4] = {{}, {}, {}, {}};
#pragma unroll 1
    for (int c = 0; c < 12; ++c) {
      const unsigned short* pb = yst + ch * MS + c * 32 + 8 * lh;
      v16bf bf = mkfrag(*(const v8us*)pb, *(const v8us*)(pb + 16));
#pragma unroll
      for (int u = 0; u < 4; ++u) {
        int row = (g0 + u) * 16 + l15;
        const unsigned short* ap = adjp + (size_t)row * NP + c * 32 + 8 * lh;
        v16bf a = mkfrag(*(const v8us*)ap, *(const v8us*)(ap + 16));
        acc[u] = wmma_bf(a, bf, acc[u]);
      }
    }
#pragma unroll
    for (int u = 0; u < 4; ++u)
#pragma unroll
      for (int r = 0; r < 8; ++r) {
        int node = (g0 + u) * 16 + r + 8 * lh;
        if (node < N) gdst[(size_t)node * D + ch] = f2bf(acc[u][r]);
      }
  }
}

// ---------------------------------------------------------------------------
// Kernel 4: out = x1 + relu(G @ theta). All 8 waves share the same A rows, so
// the G slab is staged ONCE per block into LDS via async global->LDS DMA
// (padded stride for conflict-free ds_load_b128 A-frags), then consumed by
// WMMA. Falls back to a synchronous staged copy if the builtin is absent.
// ---------------------------------------------------------------------------
__global__ void __launch_bounds__(256)
gcn2_kernel(const unsigned short* __restrict__ gbuf,
            const unsigned short* __restrict__ thp,
            const float* __restrict__ x1, float* __restrict__ out) {
  __shared__ unsigned short gsl[NP * GS];    // staged G slab [node][channel]
  const int tid = threadIdx.x, lane = tid & 31, wave = tid >> 5;
  const int lh = lane >> 4, l15 = lane & 15;
  const int bt = blockIdx.x, b = bt / T, t = bt % T;
  const unsigned short* gsrc = gbuf + (size_t)bt * NP * D;

#ifdef HAVE_ASYNC_LDS
  for (int i = tid; i < NP * 16; i += 256) {
    int row = i >> 4, c8 = (i & 15) * 8;
    gv4i_ptr gp = (gv4i_ptr)(gsrc + (size_t)row * D + c8);
    lv4i_ptr lp = (lv4i_ptr)(gsl + row * GS + c8);
    __builtin_amdgcn_global_load_async_to_lds_b128(gp, lp, 0, 0);
  }
  async_wait0();
#else
  for (int i = tid; i < NP * 16; i += 256) {
    int row = i >> 4, c8 = (i & 15) * 8;
    *(v8us*)(gsl + row * GS + c8) = *(const v8us*)(gsrc + (size_t)row * D + c8);
  }
#endif
  __syncthreads();

  const int nt = wave, ocol = nt * 16 + l15;
  for (int g0 = 0; g0 < 24; g0 += 4) {
    v8f acc[4] = {{}, {}, {}, {}};
#pragma unroll
    for (int c = 0; c < 4; ++c) {
      const unsigned short* wp = thp + ocol * 128 + c * 32 + 8 * lh;
      v16bf bf = mkfrag(*(const v8us*)wp, *(const v8us*)(wp + 16));
#pragma unroll
      for (int u = 0; u < 4; ++u) {
        int row = (g0 + u) * 16 + l15;
        const unsigned short* ap = gsl + row * GS + c * 32 + 8 * lh;
        v16bf a = mkfrag(*(const v8us*)ap, *(const v8us*)(ap + 16));
        acc[u] = wmma_bf(a, bf, acc[u]);
      }
    }
#pragma unroll
    for (int u = 0; u < 4; ++u)
#pragma unroll
      for (int r = 0; r < 8; ++r) {
        int node = (g0 + u) * 16 + r + 8 * lh;
        if (node < N) {
          size_t idx = ((size_t)(b * N + node) * T + t) * D + ocol;
          float gg = acc[u][r];
          out[idx] = x1[idx] + (gg > 0.f ? gg : 0.f);
        }
      }
  }
}

// ---------------------------------------------------------------------------
extern "C" void kernel_launch(void* const* d_in, const int* in_sizes, int n_in,
                              void* d_out, int out_size, void* d_ws, size_t ws_size,
                              hipStream_t stream) {
  (void)in_sizes; (void)n_in; (void)out_size; (void)ws_size;
  const float* x   = (const float*)d_in[0];
  const float* cqw = (const float*)d_in[1];
  const float* cqb = (const float*)d_in[2];
  const float* ckw = (const float*)d_in[3];
  const float* ckb = (const float*)d_in[4];
  const float* fcw = (const float*)d_in[5];
  const float* fcb = (const float*)d_in[6];
  const float* pjw = (const float*)d_in[7];
  const float* pjb = (const float*)d_in[8];
  const float* thw = (const float*)d_in[9];
  const float* adj = (const float*)d_in[10];
  const float* l1g = (const float*)d_in[11];
  const float* l1b = (const float*)d_in[12];
  const float* l2g = (const float*)d_in[13];
  const float* l2b = (const float*)d_in[14];

  char* ws = (char*)d_ws;
  unsigned short* wqp  = (unsigned short*)(ws + OFF_WQP);
  unsigned short* wkp  = (unsigned short*)(ws + OFF_WKP);
  unsigned short* fcp  = (unsigned short*)(ws + OFF_FCP);
  unsigned short* pjp  = (unsigned short*)(ws + OFF_PJP);
  unsigned short* thp  = (unsigned short*)(ws + OFF_THP);
  unsigned short* adjp = (unsigned short*)(ws + OFF_ADJ);
  unsigned short* y2t  = (unsigned short*)(ws + OFF_Y2);
  unsigned short* gbuf = (unsigned short*)(ws + OFF_GB);
  float* x1 = (float*)d_out;   // reuse output buffer for sublayer-1 result

  pack_kernel<<<(NP * NP + 255) / 256, 256, 0, stream>>>(
      cqw, ckw, fcw, pjw, thw, adj, wqp, wkp, fcp, pjp, thp, adjp);
  sublayer1_kernel<<<B * N, 256, 0, stream>>>(
      x, cqb, ckb, fcb, pjb, l1g, l1b, wqp, wkp, fcp, pjp, x1);
  ln2_kernel<<<(B * N * T + 7) / 8, 256, 0, stream>>>(x1, l2g, l2b, y2t);
  gcn1_kernel<<<B * T, 256, 0, stream>>>(adjp, y2t, gbuf);
  gcn2_kernel<<<B * T, 256, 0, stream>>>(gbuf, thp, x1, (float*)d_out);
}
